// AdaptiveHomeostasisTwoLayers_48026324304236
// MI455X (gfx1250) — matrix operand order repaired and checked by reading, not compiled, not run on hardware
//
#include <hip/hip_runtime.h>

// ---------------------------------------------------------------------------
// SNN forward (2-layer LIF) for MI455X / gfx1250.
//   GEMM1: cur1[T*B,H] = X[T*B,I] @ W1^T   (bf16x2 split, fp32-accum WMMA)
//   scan1: LIF recurrence in place -> spk1_rec (in d_out)
//   GEMM2: cur2[T*B,O] = spk1_rec @ W2^T   (A exact in bf16; W split)
//   scan2: LIF recurrence in place -> spk2_rec (in d_out)
// A tiles are staged into LDS by the Tensor Data Mover (double-buffered,
// hardware row padding) when the TDM builtin is available; otherwise a
// synchronous cooperative copy is used.
// ---------------------------------------------------------------------------

typedef __attribute__((ext_vector_type(16))) __bf16 v16bf;
typedef __attribute__((ext_vector_type(8)))  float  v8f;
typedef __attribute__((ext_vector_type(4)))  float  f4;
typedef __attribute__((ext_vector_type(4)))  unsigned int v4u;
typedef __attribute__((ext_vector_type(8)))  int    v8i;
typedef __attribute__((ext_vector_type(4)))  int    v4i;

#if __has_builtin(__builtin_amdgcn_tensor_load_to_lds) && \
    __has_builtin(__builtin_amdgcn_s_wait_tensorcnt)
#define USE_TDM 1
#else
#define USE_TDM 0
#endif

static constexpr int DT = 256, DB = 128, DI = 1024, DH = 1024, DO = 256;
static constexpr float BETA = 0.9f;

__device__ inline v8f zero8() {
  v8f z;
#pragma unroll
  for (int i = 0; i < 8; ++i) z[i] = 0.0f;
  return z;
}

// Split fp32 -> bf16 hi + bf16 lo residual (RNE both times): hi+lo carries
// ~16-17 mantissa bits, near-fp32 for dot products.
__global__ void split_bf16_kernel(const float* __restrict__ w,
                                  __bf16* __restrict__ hi,
                                  __bf16* __restrict__ lo, int n) {
  int i = blockIdx.x * blockDim.x + threadIdx.x;
  if (i < n) {
    float f = w[i];
    __bf16 h = (__bf16)f;
    hi[i] = h;
    lo[i] = (__bf16)(f - (float)h);
  }
}

#if USE_TDM
// Issue one TDM 2-D tile load: 128 rows x 32 fp32, global row stride K
// elements, into LDS at ldsByteAddr with 4 DWORDs of padding every 32 DWORDs
// (row stride 36 floats -> conflict-free fragment reads).
__device__ inline void tdm_load_tile(const float* gsrc, unsigned ldsByteAddr,
                                     int M, int K) {
  unsigned long long ga = (unsigned long long)(uintptr_t)(const void*)gsrc;
  v4u g0;
  g0[0] = 1u;                                    // count=1, user descriptor
  g0[1] = ldsByteAddr;                           // lds_addr (bytes)
  g0[2] = (unsigned)(ga & 0xFFFFFFFFu);          // global_addr[31:0]
  g0[3] = (unsigned)((ga >> 32) & 0x01FFFFFFu)   // global_addr[56:32]
          | (2u << 30);                          // type=2 ("image")
  v8i g1;
  g1[0] = (int)((2u << 16)      // data_size = 4 bytes
                | (1u << 20)    // pad_enable
                | (4u << 22)    // pad_interval: 2^(4+1)=32 DWORDs
                | (3u << 25));  // pad_amount: 3+1 = 4 DWORDs
  g1[1] = (int)(((unsigned)K & 0xFFFFu) << 16);            // tensor_dim0 lo
  g1[2] = (int)(((unsigned)K >> 16) |
                (((unsigned)M & 0xFFFFu) << 16));          // dim0 hi | dim1 lo
  g1[3] = (int)(((unsigned)M >> 16) | (32u << 16));        // dim1 hi | tile_dim0=32
  g1[4] = 128;                                             // tile_dim1=128, tile_dim2=0
  g1[5] = K;                                               // tensor_dim0_stride lo
  g1[6] = 0;                                               // stride hi | dim1_stride lo
  g1[7] = 0;
  v4i gz = {0, 0, 0, 0};        // 2-D tensor: groups 2/3 unused
  v8i g4 = {0, 0, 0, 0, 0, 0, 0, 0};  // unused trailing operand (VADDR4)
  __builtin_amdgcn_tensor_load_to_lds(g0, g1, gz, gz, g4, 0);
}
#endif

// C[M,N] = A[M,K] @ W[N,K]^T  with W pre-split into bf16 hi/lo.
// SPLIT_A: also split A into hi/lo (3 WMMAs/tile); else A is exact in bf16
// (2 WMMAs/tile). Block tile 128x128, 8 waves (4 along M x 2 along N), each
// wave owns a 32x64 sub-tile = 2x4 accumulators of 16x16 fp32.
template <bool SPLIT_A>
__global__ __launch_bounds__(256) void snn_gemm_bf16x2(
    const float* __restrict__ A, const __bf16* __restrict__ Whi,
    const __bf16* __restrict__ Wlo, float* __restrict__ C, int M, int N,
    int K) {
  constexpr int BK = 32;
  constexpr int LDA = 36;  // padded fp32 LDS row stride (16B aligned,
                           // conflict-free across the 16 fragment rows)
  __shared__ float As[2][128 * LDA];

  const int tid = threadIdx.x;
  const int lane = tid & 31;
  const int lrow = lane & 15;  // row (A) / col (B,C) within 16
  const int kh = lane >> 4;    // which K-half this lane holds
  const int wave = tid >> 5;

  const int mBase = blockIdx.y * 128;
  const int nBase = blockIdx.x * 128;
  const int wM = (wave & 3) * 32;
  const int wN = (wave >> 2) * 64;

  v8f acc[2][4];
#pragma unroll
  for (int mt = 0; mt < 2; ++mt)
#pragma unroll
    for (int nt = 0; nt < 4; ++nt) acc[mt][nt] = zero8();

#if USE_TDM
  const unsigned ldsB0 = (unsigned)(uintptr_t)(void*)&As[0][0];
  const unsigned ldsB1 = (unsigned)(uintptr_t)(void*)&As[1][0];
  // preload first tile via Tensor Data Mover (wave 0 drives the DMA)
  if (wave == 0) {
    tdm_load_tile(A + (size_t)mBase * K, ldsB0, M, K);
    __builtin_amdgcn_s_wait_tensorcnt(0);
  }
  __syncthreads();
#else
  const int lr = tid >> 3;       // cooperative loader: row group 0..31
  const int lc = (tid & 7) * 4;  // float4 column
#endif

  int cur = 0;
  for (int k0 = 0; k0 < K; k0 += BK) {
#if USE_TDM
    // kick off DMA of the next tile into the other buffer
    if (wave == 0 && k0 + BK < K) {
      tdm_load_tile(A + (size_t)mBase * K + (k0 + BK),
                    (cur == 0) ? ldsB1 : ldsB0, M, K);
    }
#else
    // synchronous fallback: stage A tile (128 x 32 fp32) into LDS
#pragma unroll
    for (int r = 0; r < 4; ++r) {
      const int row = r * 32 + lr;
      const float* gp = A + (size_t)(mBase + row) * K + k0 + lc;
      f4 v = *reinterpret_cast<const f4*>(gp);
      *reinterpret_cast<f4*>(&As[0][row * LDA + lc]) = v;
      if (k0 + BK < K) __builtin_prefetch(gp + BK, 0, 1);
    }
    __syncthreads();
#endif

    const float* Atile = &As[USE_TDM ? cur : 0][0];

    // ---- build A fragments per 16-bit A layout (16x32, wave32) ----
    // lanes 0-15 hold K = {c..c+7, c+16..c+23} with c=0; lanes 16-31: c=8.
    v16bf ah[2], al[2];
#pragma unroll
    for (int mt = 0; mt < 2; ++mt) {
      const float* ap = Atile + (wM + mt * 16 + lrow) * LDA + kh * 8;
      f4 f0 = *reinterpret_cast<const f4*>(ap);
      f4 f1 = *reinterpret_cast<const f4*>(ap + 4);
      f4 f2 = *reinterpret_cast<const f4*>(ap + 16);
      f4 f3 = *reinterpret_cast<const f4*>(ap + 20);
      float fv[16];
#pragma unroll
      for (int e = 0; e < 4; ++e) {
        fv[e] = f0[e];
        fv[4 + e] = f1[e];
        fv[8 + e] = f2[e];
        fv[12 + e] = f3[e];
      }
#pragma unroll
      for (int e = 0; e < 16; ++e) {
        float f = fv[e];
        __bf16 h = (__bf16)f;
        ah[mt][e] = h;
        if (SPLIT_A) al[mt][e] = (__bf16)(f - (float)h);
      }
    }

    // ---- B fragments (32x16, same K order as A) + WMMA ----
    // lane holds bf16 of column n: K = {c..c+7, c+16..c+23}, c = kh*8
    // -> two 16B loads from W[n*K + ...].
#pragma unroll
    for (int nt = 0; nt < 4; ++nt) {
      const size_t boff =
          (size_t)(nBase + wN + nt * 16 + lrow) * K + k0 + kh * 8;
      typedef __attribute__((ext_vector_type(8))) __bf16 v8bf;
      v8bf h0 = *reinterpret_cast<const v8bf*>(Whi + boff);
      v8bf h1 = *reinterpret_cast<const v8bf*>(Whi + boff + 16);
      v8bf l0 = *reinterpret_cast<const v8bf*>(Wlo + boff);
      v8bf l1 = *reinterpret_cast<const v8bf*>(Wlo + boff + 16);
      v16bf bh = __builtin_shufflevector(h0, h1, 0, 1, 2, 3, 4, 5, 6, 7, 8, 9, 10, 11, 12, 13, 14, 15);
      v16bf bl = __builtin_shufflevector(l0, l1, 0, 1, 2, 3, 4, 5, 6, 7, 8, 9, 10, 11, 12, 13, 14, 15);
#pragma unroll
      for (int mt = 0; mt < 2; ++mt) {
        acc[mt][nt] = __builtin_amdgcn_wmma_f32_16x16x32_bf16(
            false, ah[mt], false, bh, (short)0, acc[mt][nt], false, false);
        acc[mt][nt] = __builtin_amdgcn_wmma_f32_16x16x32_bf16(
            false, ah[mt], false, bl, (short)0, acc[mt][nt], false, false);
        if (SPLIT_A)
          acc[mt][nt] = __builtin_amdgcn_wmma_f32_16x16x32_bf16(
              false, al[mt], false, bh, (short)0, acc[mt][nt], false, false);
      }
    }

#if USE_TDM
    // make the next tile visible to everyone before the buffer swap
    if (wave == 0 && k0 + BK < K) __builtin_amdgcn_s_wait_tensorcnt(0);
    __syncthreads();
    cur ^= 1;
#else
    __syncthreads();
#endif
  }

  // ---- epilogue: 16x16 fp32 C/D layout: VGPR r -> M=r (lanes 0-15),
  // M=r+8 (lanes 16-31); N = lane & 15 ----
#pragma unroll
  for (int mt = 0; mt < 2; ++mt)
#pragma unroll
    for (int nt = 0; nt < 4; ++nt) {
      const int col = nBase + wN + nt * 16 + lrow;
#pragma unroll
      for (int r = 0; r < 8; ++r) {
        const int row = mBase + wM + mt * 16 + kh * 8 + r;
        C[(size_t)row * N + col] = acc[mt][nt][r];
      }
    }
}

// In-place LIF scan over time. buf holds cur[t] on entry, spk[t] on exit.
// mem = beta*mem + (cur + bias) - spk_prev*thr ; spk = (mem - thr) > 0
__global__ void lif_scan_kernel(float* __restrict__ buf,
                                const float* __restrict__ bias,
                                const float* __restrict__ thr_p, int BD,
                                int D, int Tn) {
  int i = blockIdx.x * blockDim.x + threadIdx.x;
  if (i >= BD) return;
  const float thr = *thr_p;
  const float b = bias[i % D];
  float mem = 0.0f, spk = 0.0f;
  for (int t = 0; t < Tn; ++t) {
    float* p = buf + (size_t)t * BD + i;
    float cur = *p + b;
    mem = BETA * mem + cur - spk * thr;
    spk = (mem - thr) > 0.0f ? 1.0f : 0.0f;
    *p = spk;
  }
}

extern "C" void kernel_launch(void* const* d_in, const int* in_sizes, int n_in,
                              void* d_out, int out_size, void* d_ws,
                              size_t ws_size, hipStream_t stream) {
  (void)in_sizes; (void)n_in; (void)out_size; (void)ws_size;
  const float* x    = (const float*)d_in[0];  // [T,B,I]
  const float* w1   = (const float*)d_in[1];  // [H,I]
  const float* b1   = (const float*)d_in[2];  // [H]
  const float* w2   = (const float*)d_in[3];  // [O,H]
  const float* b2   = (const float*)d_in[4];  // [O]
  const float* thr1 = (const float*)d_in[5];  // scalar
  const float* thr2 = (const float*)d_in[6];  // scalar

  float* spk1 = (float*)d_out;                // [T*B, H] cur1 -> spk1
  float* cur2 = spk1 + (size_t)DT * DB * DH;  // [T*B, O] cur2 -> spk2

  __bf16* w1h = (__bf16*)d_ws;                // 2 MB
  __bf16* w1l = w1h + (size_t)DH * DI;        // 2 MB
  __bf16* w2h = w1l + (size_t)DH * DI;        // 512 KB
  __bf16* w2l = w2h + (size_t)DO * DH;        // 512 KB   (5 MB total)

  // 1) split weights into bf16 hi/lo
  split_bf16_kernel<<<(DH * DI + 255) / 256, 256, 0, stream>>>(w1, w1h, w1l,
                                                               DH * DI);
  split_bf16_kernel<<<(DO * DH + 255) / 256, 256, 0, stream>>>(w2, w2h, w2l,
                                                               DO * DH);

  // 2) GEMM1 (A split: x is fp32-valued), cur1 into spk1_rec region
  snn_gemm_bf16x2<true><<<dim3(DH / 128, (DT * DB) / 128), 256, 0, stream>>>(
      x, w1h, w1l, spk1, DT * DB, DH, DI);

  // 3) LIF scan layer 1 (in place, bias folded in)
  lif_scan_kernel<<<(DB * DH + 255) / 256, 256, 0, stream>>>(spk1, b1, thr1,
                                                             DB * DH, DH, DT);

  // 4) GEMM2 (A = spikes, exact in bf16), cur2 into spk2_rec region
  snn_gemm_bf16x2<false><<<dim3(DO / 128, (DT * DB) / 128), 256, 0, stream>>>(
      spk1, w2h, w2l, cur2, DT * DB, DO, DH);

  // 5) LIF scan layer 2
  lif_scan_kernel<<<(DB * DO + 255) / 256, 256, 0, stream>>>(cur2, b2, thr2,
                                                             DB * DO, DO, DT);
}
